// R_HGT_60387240181860
// MI455X (gfx1250) — compile-verified, matrix operand order checked by reading, not run on hardware
//
#include <hip/hip_runtime.h>
#include <hip/hip_bf16.h>
#include <stdint.h>

// Problem constants (match reference)
#define N_NODES 60000
#define N_EDGES 120000
#define R_REL   3
#define IN_DIM  256
#define H_HEADS 8
#define DHD     32
#define D_DIM   256
#define RIN_DIM 64
#define RH_DIM  32
#define SLOPEF  0.2f

typedef __attribute__((ext_vector_type(16))) __bf16 v16bf;
typedef __attribute__((ext_vector_type(8)))  __bf16 v8bf;
typedef __attribute__((ext_vector_type(4)))  __bf16 v4bf;
typedef __attribute__((ext_vector_type(8)))  float  v8f;
typedef __attribute__((ext_vector_type(4)))  float  v4f;

__device__ __forceinline__ __bf16 f2bf(float f) {
    unsigned u = __builtin_bit_cast(unsigned, f);
    unsigned r = u + 0x7FFFu + ((u >> 16) & 1u);   // round-to-nearest-even
    unsigned short h = (unsigned short)(r >> 16);
    return __builtin_bit_cast(__bf16, h);
}
__device__ __forceinline__ float lrelu(float x) { return x > 0.f ? x : SLOPEF * x; }

// monotone float<->uint encoding so atomicMax(uint) == float max; key 0 == -inf
__device__ __forceinline__ unsigned enc_ord(float f) {
    unsigned u = __builtin_bit_cast(unsigned, f);
    return (u & 0x80000000u) ? ~u : (u | 0x80000000u);
}
__device__ __forceinline__ float dec_ord(unsigned k) {
    if (k == 0u) return 0.0f;                      // empty segment -> 0 (matches reference)
    unsigned u = (k & 0x80000000u) ? (k & 0x7FFFFFFFu) : ~k;
    return __builtin_bit_cast(float, u);
}

// ---------------------------------------------------------------------------
// Generalized batched GEMM:  C[z] = A[z] * B[z] (+ bias[z]); A f32 -> bf16 WMMA
// z decomposed as z1 = z / zdiv, z2 = z % zdiv; each operand offset z1*S1+z2*S2.
// Block: 256 thr = 8 waves (2x4), block tile 64x128, wave tile 32x32, K step 32.
// Software-pipelined: global loads for tile k+1 are issued before the WMMAs of
// tile k so HBM latency hides behind ds_load_b128 + v_wmma; staging itself is
// branch-free (clamped addresses + select-to-zero) and vectorized.
// ---------------------------------------------------------------------------
#define BM 64
#define BN 128
#define BK 32
#define GTHREADS 256
#define LDP 40   // BK + 8 pad -> row stride 80B (16B aligned fragments)

__global__ __launch_bounds__(GTHREADS)
void gemm_bf16_wmma(const float* __restrict__ A, const float* __restrict__ B,
                    const float* __restrict__ bias, float* __restrict__ C,
                    int M, int K, int Ncols, int lda, int ldb, int ldc,
                    int zdiv,
                    long aS1, long aS2, long bS1, long bS2,
                    long biasS1, long biasS2, long cS1, long cS2) {
    __shared__ __bf16 As[BM][LDP];
    __shared__ __bf16 Bs[BN][LDP];     // stored transposed: Bs[n][k]

    const int z  = blockIdx.z;
    const int z1 = z / zdiv, z2 = z - z1 * zdiv;
    A += (long)z1 * aS1 + (long)z2 * aS2;
    B += (long)z1 * bS1 + (long)z2 * bS2;
    C += (long)z1 * cS1 + (long)z2 * cS2;
    if (bias) bias += (long)z1 * biasS1 + (long)z2 * biasS2;

    const int tid  = threadIdx.x;
    const int lane = tid & 31;
    const int wave = tid >> 5;
    const int wm   = wave >> 2;            // 0..1
    const int wn   = wave & 3;             // 0..3
    const int rowBlock = blockIdx.y * BM;
    const int nBase    = blockIdx.x * BN;

    v8f acc[2][2] = {};

    const int lr = lane & 15;
    const int kb = (lane < 16) ? 0 : 8;    // per-lane K sub-block base (ISA A/B layout)

    // ---- loop-invariant staging coordinates ----
    // A: 512 float4-quads per tile, 2 per thread.
    int qa_r[2], qa_c4[2]; long qa_off[2]; bool okA[2];
    #pragma unroll
    for (int i = 0; i < 2; ++i) {
        int q = i * GTHREADS + tid;
        qa_r[i]  = q >> 3;
        qa_c4[i] = (q & 7) * 4;
        int gr = rowBlock + qa_r[i];
        okA[i] = gr < M;
        int grc = okA[i] ? gr : (M - 1);
        qa_off[i] = (long)grc * lda + qa_c4[i];
    }
    // B: 1024 4-deep k-strips per tile, 4 per thread.
    int qb_c[4], qb_k4[4]; long qb_off[4]; bool okB[4];
    #pragma unroll
    for (int i = 0; i < 4; ++i) {
        int q = i * GTHREADS + tid;
        qb_c[i]  = q & (BN - 1);
        qb_k4[i] = (q >> 7) * 4;
        int gc = nBase + qb_c[i];
        okB[i] = gc < Ncols;
        int gcc = okB[i] ? gc : (Ncols - 1);
        qb_off[i] = (long)qb_k4[i] * ldb + gcc;
    }

    // ---- prologue: load tile k=0 into registers ----
    v4f aR[2]; float bR[4][4];
    #pragma unroll
    for (int i = 0; i < 2; ++i)
        aR[i] = *reinterpret_cast<const v4f*>(A + qa_off[i]);
    #pragma unroll
    for (int i = 0; i < 4; ++i)
        #pragma unroll
        for (int j = 0; j < 4; ++j)
            bR[i][j] = B[qb_off[i] + (long)j * ldb];

    for (int k0 = 0; k0 < K; k0 += BK) {
        // ---- convert + commit current registers to LDS ----
        #pragma unroll
        for (int i = 0; i < 2; ++i) {
            v4bf o;
            #pragma unroll
            for (int j = 0; j < 4; ++j) o[j] = f2bf(okA[i] ? aR[i][j] : 0.0f);
            *reinterpret_cast<v4bf*>(&As[qa_r[i]][qa_c4[i]]) = o;
        }
        #pragma unroll
        for (int i = 0; i < 4; ++i) {
            v4bf o;
            #pragma unroll
            for (int j = 0; j < 4; ++j) o[j] = f2bf(okB[i] ? bR[i][j] : 0.0f);
            *reinterpret_cast<v4bf*>(&Bs[qb_c[i]][qb_k4[i]]) = o;
        }
        __syncthreads();

        // ---- issue global loads for the NEXT tile (latency hidden by WMMAs) ----
        const int kn = k0 + BK;
        const bool more = kn < K;
        if (more) {
            #pragma unroll
            for (int i = 0; i < 2; ++i)
                aR[i] = *reinterpret_cast<const v4f*>(A + qa_off[i] + kn);
            #pragma unroll
            for (int i = 0; i < 4; ++i)
                #pragma unroll
                for (int j = 0; j < 4; ++j)
                    bR[i][j] = B[qb_off[i] + (long)(kn + j) * ldb];
            if (kn + BK < K) {
                #pragma unroll
                for (int i = 0; i < 2; ++i)
                    __builtin_prefetch(A + qa_off[i] + kn + BK, 0, 1);
                #pragma unroll
                for (int i = 0; i < 4; ++i)
                    __builtin_prefetch(B + qb_off[i] + (long)(kn + BK) * ldb, 0, 1);
            }
        }

        // ---- fragments + WMMA on current tile ----
        v16bf af[2], bfv[2];
        #pragma unroll
        for (int mi = 0; mi < 2; ++mi) {
            const __bf16* p = &As[wm * 32 + mi * 16 + lr][0];
            v8bf lo = *reinterpret_cast<const v8bf*>(p + kb);
            v8bf hi = *reinterpret_cast<const v8bf*>(p + kb + 16);
            af[mi] = __builtin_shufflevector(lo, hi, 0,1,2,3,4,5,6,7,8,9,10,11,12,13,14,15);
        }
        #pragma unroll
        for (int ni = 0; ni < 2; ++ni) {
            const __bf16* p = &Bs[wn * 32 + ni * 16 + lr][0];
            v8bf lo = *reinterpret_cast<const v8bf*>(p + kb);
            v8bf hi = *reinterpret_cast<const v8bf*>(p + kb + 16);
            bfv[ni] = __builtin_shufflevector(lo, hi, 0,1,2,3,4,5,6,7,8,9,10,11,12,13,14,15);
        }
        #pragma unroll
        for (int mi = 0; mi < 2; ++mi)
            #pragma unroll
            for (int ni = 0; ni < 2; ++ni)
                acc[mi][ni] = __builtin_amdgcn_wmma_f32_16x16x32_bf16(
                    false, af[mi], false, bfv[ni], (short)0, acc[mi][ni], false, false);
        __syncthreads();
    }

    // ---- store: VGPR j holds M=j (lanes 0-15) / M=j+8 (lanes 16-31), N=lane%16 ----
    #pragma unroll
    for (int mi = 0; mi < 2; ++mi) {
        #pragma unroll
        for (int ni = 0; ni < 2; ++ni) {
            int col = nBase + wn * 32 + ni * 16 + lr;
            if (col >= Ncols) continue;
            float bv = bias ? bias[col] : 0.0f;
            int rbase = rowBlock + wm * 32 + mi * 16 + ((lane >= 16) ? 8 : 0);
            #pragma unroll
            for (int j = 0; j < 8; ++j) {
                int row = rbase + j;
                if (row < M) C[(long)row * ldc + col] = acc[mi][ni][j] + bv;
            }
        }
    }
}

// ---------------------------------------------------------------------------
// rw[r] = rel_emb[r] @ Wrel[r]   -> (R, H*2*DH) = (R, 512)
// ---------------------------------------------------------------------------
__global__ void relw_kernel(const float* __restrict__ rel_emb, const float* __restrict__ Wrel,
                            float* __restrict__ rw) {
    int t = blockIdx.x * blockDim.x + threadIdx.x;
    if (t >= R_REL * 512) return;
    int r = t / 512, o = t - r * 512;
    float s = 0.f;
    for (int i = 0; i < RIN_DIM; ++i)
        s += rel_emb[r * RIN_DIM + i] * Wrel[((long)r * RIN_DIM + i) * 512 + o];
    rw[t] = s;
}

// e_src[r,n,h] = <fs[r,n,h,:], rw[r,h,:DH]> ; e_dst[r,n,h] = <feat[n,h,:], rw[r,h,DH:]>
__global__ void escore_kernel(const float* __restrict__ fs, const float* __restrict__ feat,
                              const float* __restrict__ rw,
                              float* __restrict__ esrc, float* __restrict__ edst) {
    long t = blockIdx.x * (long)blockDim.x + threadIdx.x;
    const long total = (long)R_REL * N_NODES * H_HEADS;
    if (t >= total) return;
    int h = (int)(t % H_HEADS);
    long nr = t / H_HEADS;
    int n = (int)(nr % N_NODES);
    int r = (int)(nr / N_NODES);
    const float* fsp = fs + ((long)r * N_NODES + n) * D_DIM + h * DHD;
    const float* fp  = feat + (long)n * D_DIM + h * DHD;
    const float* rws = rw + r * 512 + h * 64;
    float s1 = 0.f, s2 = 0.f;
    #pragma unroll
    for (int d = 0; d < DHD; ++d) { s1 += fsp[d] * rws[d]; s2 += fp[d] * rws[DHD + d]; }
    esrc[t] = s1; edst[t] = s2;
}

// per-edge logits + segment max over dst (ordered-uint atomic max)
__global__ void logits_kernel(const int* __restrict__ src, const int* __restrict__ dst,
                              const float* __restrict__ esrc, const float* __restrict__ edst,
                              float* __restrict__ logits, unsigned* __restrict__ mx) {
    long t = blockIdx.x * (long)blockDim.x + threadIdx.x;
    const long total = (long)R_REL * N_EDGES;
    if (t >= total) return;
    int r = (int)(t / N_EDGES);
    int u = src[t], v = dst[t];
    const float* es = esrc + ((long)r * N_NODES + u) * H_HEADS;
    const float* ed = edst + ((long)r * N_NODES + v) * H_HEADS;
    unsigned* mxp = mx + ((long)r * N_NODES + v) * H_HEADS;
    #pragma unroll
    for (int h = 0; h < H_HEADS; ++h) {
        float lg = lrelu(es[h] + ed[h]);
        logits[t * H_HEADS + h] = lg;
        atomicMax(&mxp[h], enc_ord(lg));
    }
}

// ex = exp(logit - m); den[v] += ex  (ex overwrites logits buffer)
__global__ void expsum_kernel(const int* __restrict__ dst, float* __restrict__ logits,
                              const unsigned* __restrict__ mx, float* __restrict__ den) {
    long t = blockIdx.x * (long)blockDim.x + threadIdx.x;
    const long total = (long)R_REL * N_EDGES;
    if (t >= total) return;
    int r = (int)(t / N_EDGES);
    int v = dst[t];
    const unsigned* mxp = mx + ((long)r * N_NODES + v) * H_HEADS;
    float* dn = den + ((long)r * N_NODES + v) * H_HEADS;
    #pragma unroll
    for (int h = 0; h < H_HEADS; ++h) {
        float ex = __expf(logits[t * H_HEADS + h] - dec_ord(mxp[h]));
        logits[t * H_HEADS + h] = ex;
        atomicAdd(&dn[h], ex);
    }
}

// aggr[v] += fs[u] * alpha   (one thread per (r,edge,head))
__global__ void aggregate_kernel(const int* __restrict__ src, const int* __restrict__ dst,
                                 const float* __restrict__ logits, const float* __restrict__ den,
                                 const float* __restrict__ fs, float* __restrict__ aggr) {
    long t = blockIdx.x * (long)blockDim.x + threadIdx.x;
    const long total = (long)R_REL * N_EDGES * H_HEADS;
    if (t >= total) return;
    int h = (int)(t % H_HEADS);
    long e2 = t / H_HEADS;
    int r = (int)(e2 / N_EDGES);
    long eidx = e2;                       // == r*E + j
    int u = src[eidx], v = dst[eidx];
    float alpha = logits[eidx * H_HEADS + h] /
                  (den[((long)r * N_NODES + v) * H_HEADS + h] + 1e-16f);
    const float* fsp = fs + ((long)r * N_NODES + u) * D_DIM + h * DHD;
    float* ap = aggr + ((long)r * N_NODES + v) * D_DIM + h * DHD;
    #pragma unroll
    for (int d = 0; d < DHD; ++d) atomicAdd(&ap[d], fsp[d] * alpha);
}

// outs = a*relu(aggr) + (1-a)*res   (in place on aggr buffer)
__global__ void resid_kernel(float* __restrict__ outs, const float* __restrict__ res,
                             const float* __restrict__ res_w) {
    long t = blockIdx.x * (long)blockDim.x + threadIdx.x;
    const long total = (long)R_REL * N_NODES * D_DIM;
    if (t >= total) return;
    float a = 1.f / (1.f + __expf(-res_w[0]));
    float o = outs[t]; o = o > 0.f ? o : 0.f;
    long nd = t % ((long)N_NODES * D_DIM);
    outs[t] = a * o + (1.f - a) * res[nd];
}

// relation crossing attention: softmax over r of lrelu(<f[r,n,h,:], Wcross[q,h,:]>)
__global__ void cross_kernel(const float* __restrict__ f, const float* __restrict__ Wcross,
                             float* __restrict__ crossed) {
    long t = blockIdx.x * (long)blockDim.x + threadIdx.x;
    const long total = (long)R_REL * N_NODES * H_HEADS;   // (q, n, h)
    if (t >= total) return;
    int q = (int)(t / ((long)N_NODES * H_HEADS));
    long rem = t % ((long)N_NODES * H_HEADS);
    int n = (int)(rem / H_HEADS);
    int h = (int)(rem % H_HEADS);
    const float* wc = Wcross + ((long)q * H_HEADS + h) * DHD;
    float fr[R_REL][DHD], s[R_REL];
    #pragma unroll
    for (int r = 0; r < R_REL; ++r) {
        const float* fp = f + ((long)r * N_NODES + n) * D_DIM + h * DHD;
        float acc = 0.f;
        #pragma unroll
        for (int d = 0; d < DHD; ++d) { float v = fp[d]; fr[r][d] = v; acc += v * wc[d]; }
        s[r] = lrelu(acc);
    }
    float m = s[0];
    #pragma unroll
    for (int r = 1; r < R_REL; ++r) m = fmaxf(m, s[r]);
    float es[R_REL], sum = 0.f;
    #pragma unroll
    for (int r = 0; r < R_REL; ++r) { es[r] = __expf(s[r] - m); sum += es[r]; }
    float inv = 1.f / sum;
    float* cp = crossed + ((long)q * N_NODES + n) * D_DIM + h * DHD;
    #pragma unroll
    for (int d = 0; d < DHD; ++d) {
        float v = 0.f;
        #pragma unroll
        for (int r = 0; r < R_REL; ++r) v += es[r] * inv * fr[r][d];
        cp[d] = v;
    }
}

// rel_new = rel_emb @ Wprop + bprop   -> (R, H*RH)
__global__ void relprop_kernel(const float* __restrict__ rel_emb, const float* __restrict__ Wprop,
                               const float* __restrict__ bprop, float* __restrict__ rel_new) {
    int t = blockIdx.x * blockDim.x + threadIdx.x;
    const int HR = H_HEADS * RH_DIM;                       // 256
    if (t >= R_REL * HR) return;
    int r = t / HR, o = t - r * HR;
    float s = bprop[t];
    for (int i = 0; i < RIN_DIM; ++i)
        s += rel_emb[r * RIN_DIM + i] * Wprop[((long)r * RIN_DIM + i) * HR + o];
    rel_new[t] = s;
}

// rf[r,h,e] = sum_k rel_new[r,h,k] * Wfuse_rel[r,h,k,e]
__global__ void rf_kernel(const float* __restrict__ rel_new, const float* __restrict__ Wfr,
                          float* __restrict__ rf) {
    int t = blockIdx.x * blockDim.x + threadIdx.x;
    if (t >= R_REL * H_HEADS * DHD) return;
    int r = t / (H_HEADS * DHD);
    int rem = t - r * (H_HEADS * DHD);
    int h = rem / DHD, e = rem - h * DHD;
    float s = 0.f;
    for (int k = 0; k < RH_DIM; ++k)
        s += rel_new[r * H_HEADS * RH_DIM + h * RH_DIM + k] *
             Wfr[(((long)r * H_HEADS + h) * RH_DIM + k) * DHD + e];
    rf[t] = s;
}

// final relation fusing: softmax over r of lrelu(<nf[r,n,h,:], rf[r,h,:]>), weighted sum
__global__ void fuse_kernel(const float* __restrict__ nf, const float* __restrict__ rf,
                            float* __restrict__ out) {
    long t = blockIdx.x * (long)blockDim.x + threadIdx.x;
    const long total = (long)N_NODES * H_HEADS;
    if (t >= total) return;
    int n = (int)(t / H_HEADS);
    int h = (int)(t % H_HEADS);
    float nv[R_REL][DHD], s[R_REL];
    #pragma unroll
    for (int r = 0; r < R_REL; ++r) {
        const float* np = nf + ((long)r * N_NODES + n) * D_DIM + h * DHD;
        const float* rp = rf + ((long)r * H_HEADS + h) * DHD;
        float acc = 0.f;
        #pragma unroll
        for (int d = 0; d < DHD; ++d) { float v = np[d]; nv[r][d] = v; acc += v * rp[d]; }
        s[r] = lrelu(acc);
    }
    float m = s[0];
    #pragma unroll
    for (int r = 1; r < R_REL; ++r) m = fmaxf(m, s[r]);
    float es[R_REL], sum = 0.f;
    #pragma unroll
    for (int r = 0; r < R_REL; ++r) { es[r] = __expf(s[r] - m); sum += es[r]; }
    float inv = 1.f / sum;
    float* op = out + (long)n * D_DIM + h * DHD;
    #pragma unroll
    for (int d = 0; d < DHD; ++d) {
        float v = 0.f;
        #pragma unroll
        for (int r = 0; r < R_REL; ++r) v += es[r] * inv * nv[r][d];
        op[d] = v;
    }
}

// ---------------------------------------------------------------------------
extern "C" void kernel_launch(void* const* d_in, const int* in_sizes, int n_in,
                              void* d_out, int out_size, void* d_ws, size_t ws_size,
                              hipStream_t stream) {
    const float* x       = (const float*)d_in[0];
    const float* Wp      = (const float*)d_in[1];
    const float* bp      = (const float*)d_in[2];
    const float* Wn      = (const float*)d_in[3];
    const float* bn      = (const float*)d_in[4];
    const float* Wsrc    = (const float*)d_in[5];
    const float* bsrc    = (const float*)d_in[6];
    const float* rel_emb = (const float*)d_in[7];
    const float* Wrel    = (const float*)d_in[8];
    const float* Wres    = (const float*)d_in[9];
    const float* bres    = (const float*)d_in[10];
    const float* res_w   = (const float*)d_in[11];
    const float* Wcross  = (const float*)d_in[12];
    const float* Wprop   = (const float*)d_in[13];
    const float* bprop   = (const float*)d_in[14];
    const float* Wfn     = (const float*)d_in[15];
    const float* Wfr     = (const float*)d_in[16];
    const int*   src_idx = (const int*)d_in[17];
    const int*   dst_idx = (const int*)d_in[18];
    float* out = (float*)d_out;

    const size_t NP  = (size_t)N_NODES * D_DIM;            // 15.36M
    const size_t RNH = (size_t)R_REL * N_NODES * H_HEADS;  // 1.44M
    const size_t REH = (size_t)R_REL * N_EDGES * H_HEADS;  // 2.88M

    float* f = (float*)d_ws;
    size_t o = 0;
    float* xp      = f + o; o += NP;
    float* feat    = f + o; o += NP;
    float* fs      = f + o; o += 3 * NP;       // later reused as `crossed`
    float* res     = f + o; o += NP;
    float* outs    = f + o; o += 3 * NP;       // aggr -> outs -> nf (reused)
    float* esrc    = f + o; o += RNH;
    float* edst    = f + o; o += RNH;
    float* den     = f + o; o += RNH;
    float* logits  = f + o; o += REH;
    float* rw      = f + o; o += (size_t)R_REL * 512;
    float* rel_new = f + o; o += (size_t)R_REL * H_HEADS * RH_DIM;
    float* rfb     = f + o; o += (size_t)R_REL * H_HEADS * DHD;
    unsigned* mx   = (unsigned*)(f + o); o += RNH;
    (void)ws_size; (void)in_sizes; (void)n_in; (void)out_size;

    dim3 blk(GTHREADS);
    const int gy = (N_NODES + BM - 1) / BM;

    // 1) xp = x @ Wp + bp
    gemm_bf16_wmma<<<dim3(D_DIM / BN, gy, 1), blk, 0, stream>>>(
        x, Wp, bp, xp, N_NODES, IN_DIM, D_DIM, IN_DIM, D_DIM, D_DIM,
        1, 0, 0, 0, 0, 0, 0, 0, 0);
    // 2) feat = xp @ Wn + bn
    gemm_bf16_wmma<<<dim3(D_DIM / BN, gy, 1), blk, 0, stream>>>(
        xp, Wn, bn, feat, N_NODES, D_DIM, D_DIM, D_DIM, D_DIM, D_DIM,
        1, 0, 0, 0, 0, 0, 0, 0, 0);
    // 3) fs[r] = feat @ Wsrc[r] + bsrc[r]  (batched over R)
    gemm_bf16_wmma<<<dim3(D_DIM / BN, gy, R_REL), blk, 0, stream>>>(
        feat, Wsrc, bsrc, fs, N_NODES, D_DIM, D_DIM, D_DIM, D_DIM, D_DIM,
        1, 0, 0, (long)D_DIM * D_DIM, 0, (long)D_DIM, 0, (long)NP, 0);
    // 4) res = xp @ Wres + bres
    gemm_bf16_wmma<<<dim3(D_DIM / BN, gy, 1), blk, 0, stream>>>(
        xp, Wres, bres, res, N_NODES, D_DIM, D_DIM, D_DIM, D_DIM, D_DIM,
        1, 0, 0, 0, 0, 0, 0, 0, 0);

    // 5) rw = rel_emb @ Wrel
    relw_kernel<<<(R_REL * 512 + 255) / 256, 256, 0, stream>>>(rel_emb, Wrel, rw);
    // 6) e_src / e_dst
    escore_kernel<<<(unsigned)((RNH + 255) / 256), 256, 0, stream>>>(fs, feat, rw, esrc, edst);

    // 7) zero segment-max keys, denominators, aggregation buffer
    hipMemsetAsync(mx, 0, RNH * sizeof(unsigned), stream);
    hipMemsetAsync(den, 0, RNH * sizeof(float), stream);
    hipMemsetAsync(outs, 0, 3 * NP * sizeof(float), stream);

    const long RE = (long)R_REL * N_EDGES;
    // 8) logits + segment max
    logits_kernel<<<(unsigned)((RE + 255) / 256), 256, 0, stream>>>(
        src_idx, dst_idx, esrc, edst, logits, mx);
    // 9) exp + segment sum
    expsum_kernel<<<(unsigned)((RE + 255) / 256), 256, 0, stream>>>(dst_idx, logits, mx, den);
    // 10) weighted scatter-add aggregation
    aggregate_kernel<<<(unsigned)((RE * H_HEADS + 255) / 256), 256, 0, stream>>>(
        src_idx, dst_idx, logits, den, fs, outs);
    // 11) gated residual (in place)
    resid_kernel<<<(unsigned)((3 * NP + 255) / 256), 256, 0, stream>>>(outs, res, res_w);

    // 12) relation crossing attention: crossed (reuses fs buffer)
    cross_kernel<<<(unsigned)((RNH + 255) / 256), 256, 0, stream>>>(outs, Wcross, fs);

    // 13) relation embedding propagation + rf
    relprop_kernel<<<(R_REL * H_HEADS * RH_DIM + 255) / 256, 256, 0, stream>>>(
        rel_emb, Wprop, bprop, rel_new);
    rf_kernel<<<(R_REL * H_HEADS * DHD + 255) / 256, 256, 0, stream>>>(rel_new, Wfr, rfb);

    // 14) nf[r,h] = crossed[r,:,h,:] @ Wfuse_node[r,h]  (batched WMMA, z = r*H + h)
    gemm_bf16_wmma<<<dim3(1, gy, R_REL * H_HEADS), blk, 0, stream>>>(
        fs /*crossed*/, Wfn, nullptr, outs /*nf*/,
        N_NODES, DHD, DHD, D_DIM, DHD, D_DIM,
        H_HEADS,
        (long)NP, (long)DHD,                       // A: r-stride, h-stride
        (long)H_HEADS * DHD * DHD, (long)DHD * DHD,// B: r-stride, h-stride
        0, 0,
        (long)NP, (long)DHD);                      // C: r-stride, h-stride

    // 15) final relation fusing -> output
    fuse_kernel<<<(unsigned)(((size_t)N_NODES * H_HEADS + 255) / 256), 256, 0, stream>>>(
        outs, rfb, out);
}
